// Decoder_2542620639997
// MI455X (gfx1250) — matrix operand order repaired and checked by reading, not compiled, not run on hardware
//
#include <hip/hip_runtime.h>
#include <hip/hip_bf16.h>

// ---------------------------------------------------------------------------
// Problem constants (match reference)
// ---------------------------------------------------------------------------
#define BB 64
#define CC 200
#define HH 1024
#define EE 1024
#define VV 50257
#define H3 3072
#define NEG_INF_F (-100000.0f)

typedef __attribute__((ext_vector_type(16))) __bf16   v16bf;
typedef __attribute__((ext_vector_type(8)))  float    v8f;
typedef __attribute__((ext_vector_type(8)))  unsigned v8u;

// ---------------------------------------------------------------------------
// fp32 -> packed bf16 pair helpers
// ---------------------------------------------------------------------------
// RNE version (used in one-shot pack kernels; accuracy-preferred)
__device__ __forceinline__ unsigned pack_bf16_rne(float a, float b) {
  unsigned ua = __builtin_bit_cast(unsigned, a);
  unsigned ub = __builtin_bit_cast(unsigned, b);
  ua = (ua + 0x7FFFu + ((ua >> 16) & 1u)) >> 16;
  ub = (ub + 0x7FFFu + ((ub >> 16) & 1u)) >> 16;
  return ua | (ub << 16);
}

// Hot-loop version: hardware v_cvt_pk_bf16_f32 if the builtin exists,
// else a 2-op truncating pack (v_lshrrev + v_and_or).
#if defined(__has_builtin)
#if __has_builtin(__builtin_amdgcn_cvt_pk_bf16_f32)
#define HAVE_CVT_PK_BF16 1
#endif
#endif

__device__ __forceinline__ unsigned pack_bf16_fast(float a, float b) {
#ifdef HAVE_CVT_PK_BF16
  auto r = __builtin_amdgcn_cvt_pk_bf16_f32(a, b);
  return __builtin_bit_cast(unsigned, r);
#else
  unsigned ua = __builtin_bit_cast(unsigned, a) >> 16;
  unsigned ub = __builtin_bit_cast(unsigned, b) & 0xFFFF0000u;
  return ub | ua;
#endif
}

__device__ __forceinline__ v8f wmma_bf16(v16bf a, v16bf b, v8f c) {
  return __builtin_amdgcn_wmma_f32_16x16x32_bf16(
      /*neg_a=*/false, a, /*neg_b=*/false, b,
      /*c_mod=*/(short)0, c, /*reuse_a=*/false, /*reuse_b=*/false);
}

// ---------------------------------------------------------------------------
// Packed-tile I/O. A tile (16x32) or B tile (32x16) in bf16 occupies
// 32 lanes x 32 bytes, stored contiguously: tile index * 1KB + lane * 32B.
// K-loop over kc-minor tiles streams contiguously (perfect coalescing).
// ---------------------------------------------------------------------------
__device__ __forceinline__ v16bf load_packed(const unsigned* __restrict__ P,
                                             int tile, int lane) {
  const v8u* p = (const v8u*)(P + ((size_t)tile * 32 + lane) * 8);
  return __builtin_bit_cast(v16bf, *p);  // 2x global_load_b128
}

// ---- inline B operand from row-major N x K matrix (B(k,n) = W[n*ldw+k]) ---
// Per-lane: 16 consecutive K floats of one W row -> 4x b128 + 8 fast packs
__device__ __forceinline__ v16bf load_bt_tile(const float* Wnk, int ldw,
                                              int lane) {
  const int n = lane & 15, half = lane >> 4;
  const float* p = Wnk + (size_t)n * ldw + 16 * half;
  v8u u;
#pragma unroll
  for (int j = 0; j < 8; ++j) u[j] = pack_bf16_fast(p[2 * j], p[2 * j + 1]);
  return __builtin_bit_cast(v16bf, u);
}

// Same with per-lane column clamp for the ragged V tail (clamped columns
// load row 0; finite garbage, never stored).
__device__ __forceinline__ v16bf load_bt_clamp(const float* W, int ldw, int n0,
                                               int k0, int nmax, int lane) {
  int n = n0 + (lane & 15);
  if (n >= nmax) n = 0;
  const float* p = W + (size_t)n * ldw + k0 + 16 * (lane >> 4);
  v8u u;
#pragma unroll
  for (int j = 0; j < 8; ++j) u[j] = pack_bf16_fast(p[2 * j], p[2 * j + 1]);
  return __builtin_bit_cast(v16bf, u);
}

// ---------------------------------------------------------------------------
// Pack kernels (one wave-lane-group of 32 per tile).
// A layout (16-bit A 16x32, contiguous k order used here): row = lane&15;
//   lanes 0-15: K = 0..15; lanes 16-31: K = 16..31
// B layout: col = lane&15; lanes 0-15: K=0..15; lanes 16-31: K=16..31
// ---------------------------------------------------------------------------
__global__ void k_pack_a(const float* __restrict__ A, int K32,
                         unsigned* __restrict__ out) {
  const int tile = blockIdx.x * 8 + (threadIdx.x >> 5);  // mt*K32 + kc
  const int lane = threadIdx.x & 31;
  const int mt = tile / K32, kc = tile % K32;
  const int K = K32 * 32;
  const float* row = A + ((size_t)mt * 16 + (lane & 15)) * K + kc * 32;
  const int half = lane >> 4;
  v8u u;
#pragma unroll
  for (int j = 0; j < 8; ++j) {
    const int k = 16 * half + 2 * j;
    u[j] = pack_bf16_rne(row[k], row[k + 1]);
  }
  *(v8u*)(out + ((size_t)tile * 32 + lane) * 8) = u;
}

__global__ void k_pack_bkn(const float* __restrict__ W, int K32, int N,
                           unsigned* __restrict__ out) {
  const int tile = blockIdx.x * 8 + (threadIdx.x >> 5);  // nt*K32 + kc
  const int lane = threadIdx.x & 31;
  const int nt = tile / K32, kc = tile % K32;
  const int n = nt * 16 + (lane & 15), half = lane >> 4;
  const float* p = W + (size_t)(kc * 32 + 16 * half) * N + n;
  v8u u;
#pragma unroll
  for (int j = 0; j < 8; ++j)
    u[j] = pack_bf16_rne(p[(size_t)(2 * j) * N], p[(size_t)(2 * j + 1) * N]);
  *(v8u*)(out + ((size_t)tile * 32 + lane) * 8) = u;
}

// GRU input x = [input | topic | applied] virtual concat, packed as A.
__device__ __forceinline__ const float* xrow(const float* i0, const float* i1,
                                             const float* i2, int b, int k,
                                             int* kk) {
  if (k < HH) { *kk = k; return i0 + (size_t)b * EE; }
  if (k < 2 * HH) { *kk = k - HH; return i1 + (size_t)b * HH; }
  *kk = k - 2 * HH;
  return i2 + (size_t)b * HH;
}

__global__ void k_pack_x(const float* __restrict__ i0,
                         const float* __restrict__ i1,
                         const float* __restrict__ i2,
                         unsigned* __restrict__ out) {
  const int tile = blockIdx.x * 8 + (threadIdx.x >> 5);  // mt*96 + kc
  const int lane = threadIdx.x & 31;
  const int mt = tile / 96, kc = tile % 96;
  const int r = mt * 16 + (lane & 15), half = lane >> 4;
  v8u u;
#pragma unroll
  for (int j = 0; j < 8; ++j) {
    const int k = kc * 32 + 16 * half + 2 * j;
    int kk;
    const float* row = xrow(i0, i1, i2, r, k, &kk);  // 32-chunks never straddle
    u[j] = pack_bf16_rne(row[kk], row[kk + 1]);
  }
  *(v8u*)(out + ((size_t)tile * 32 + lane) * 8) = u;
}

// ---------------------------------------------------------------------------
// Kernel 1: q = h0 @ dyna_W. Both operands pre-packed: loop = loads + wmma.
// 64 waves (one per n-tile), 4 m-tiles each -> dyna_W streamed once.
// ---------------------------------------------------------------------------
__global__ void k_gemm_q(const unsigned* __restrict__ pA,
                         const unsigned* __restrict__ pB,
                         float* __restrict__ q) {
  const int nt = blockIdx.x * 8 + (threadIdx.x >> 5);
  const int lane = threadIdx.x & 31;
  v8f acc[4] = {};
  for (int kc = 0; kc < 32; ++kc) {
    v16bf b = load_packed(pB, nt * 32 + kc, lane);
#pragma unroll
    for (int m = 0; m < 4; ++m) {
      v16bf a = load_packed(pA, m * 32 + kc, lane);
      acc[m] = wmma_bf16(a, b, acc[m]);
    }
  }
  const int row0 = (lane >> 4) * 8, col = nt * 16 + (lane & 15);
#pragma unroll
  for (int m = 0; m < 4; ++m)
#pragma unroll
    for (int i = 0; i < 8; ++i)
      q[(size_t)(m * 16 + row0 + i) * HH + col] = acc[m][i];
}

// ---------------------------------------------------------------------------
// Kernel 2: scores[b,c] = sum_h tanh(q[b,h] + (LCH[c]@U)[b,h]) * v[h], masked
// Block = (c, 16-row b-tile): 800 blocks x 8 waves; wave owns 8 n-tiles.
// LCH and U pre-packed (both L2-resident): inner loop = 9 loads + 8 wmma.
// ---------------------------------------------------------------------------
__global__ void k_scores(const unsigned* __restrict__ pA,
                         const unsigned* __restrict__ pU,
                         const float* __restrict__ q,
                         const float* __restrict__ dv,
                         const unsigned char* __restrict__ pad,
                         float* __restrict__ scores) {
  __shared__ float ssum[16];
  const int c = blockIdx.x >> 2;
  const int btile = blockIdx.x & 3;
  const int b0 = btile * 16;
  const int wid = threadIdx.x >> 5, lane = threadIdx.x & 31;
  if (threadIdx.x < 16) ssum[threadIdx.x] = 0.0f;
  __syncthreads();

  const int atile0 = (c * 4 + btile) * 32;
  v8f acc[8] = {};
  for (int kc = 0; kc < 32; ++kc) {
    v16bf a = load_packed(pA, atile0 + kc, lane);
#pragma unroll
    for (int t = 0; t < 8; ++t) {
      const int nt = wid + t * 8;
      v16bf b = load_packed(pU, nt * 32 + kc, lane);
      acc[t] = wmma_bf16(a, b, acc[t]);
    }
  }
  const int row0 = (lane >> 4) * 8;
#pragma unroll
  for (int t = 0; t < 8; ++t) {
    const int col = (wid + t * 8) * 16 + (lane & 15);
    const float vw = dv[col];
#pragma unroll
    for (int i = 0; i < 8; ++i) {
      const int b = b0 + row0 + i;
      const float s = tanhf(acc[t][i] + q[(size_t)b * HH + col]) * vw;
      atomicAdd(&ssum[row0 + i], s);  // ds_add_f32
    }
  }
  __syncthreads();
  if (threadIdx.x < 16) {
    const int b = b0 + threadIdx.x;
    scores[b * CC + c] = pad[b * CC + c] ? NEG_INF_F : ssum[threadIdx.x];
  }
}

// ---------------------------------------------------------------------------
// Kernel 3: softmax over C + applied[b,h] = sum_c attn[b,c]*CH[b,c,h]
// ---------------------------------------------------------------------------
__global__ void k_attn_apply(const float* __restrict__ scores,
                             const float* __restrict__ ch,
                             float* __restrict__ applied) {
  __shared__ float sattn[CC];
  __shared__ float red[256];
  const int b = blockIdx.x, tid = threadIdx.x;
  for (int c = tid; c < CC; c += 256) sattn[c] = scores[b * CC + c];
  __syncthreads();
  float lm = -3.0e38f;
  for (int c = tid; c < CC; c += 256) lm = fmaxf(lm, sattn[c]);
  red[tid] = lm;
  __syncthreads();
  for (int s = 128; s > 0; s >>= 1) {
    if (tid < s) red[tid] = fmaxf(red[tid], red[tid + s]);
    __syncthreads();
  }
  const float mx = red[0];
  __syncthreads();
  float ls = 0.0f;
  for (int c = tid; c < CC; c += 256) {
    const float e = __expf(sattn[c] - mx);
    sattn[c] = e;
    ls += e;
  }
  red[tid] = ls;
  __syncthreads();
  for (int s = 128; s > 0; s >>= 1) {
    if (tid < s) red[tid] += red[tid + s];
    __syncthreads();
  }
  const float inv = 1.0f / red[0];
  __syncthreads();
  for (int c = tid; c < CC; c += 256) sattn[c] *= inv;
  __syncthreads();
  float acc0 = 0.f, acc1 = 0.f, acc2 = 0.f, acc3 = 0.f;
  const float* base = ch + (size_t)b * CC * HH;
  for (int c = 0; c < CC; ++c) {
    const float w = sattn[c];
    const float* row = base + (size_t)c * HH;
    acc0 += w * row[tid];
    acc1 += w * row[tid + 256];
    acc2 += w * row[tid + 512];
    acc3 += w * row[tid + 768];
  }
  float* out = applied + (size_t)b * HH;
  out[tid] = acc0;
  out[tid + 256] = acc1;
  out[tid + 512] = acc2;
  out[tid + 768] = acc3;
}

// ---------------------------------------------------------------------------
// Kernel 4a: GRU double GEMM, 4 m-tiles per wave -> W_ih/W_hh streamed once.
// accA = x @ W_ih^T (K=3072), accB = h0 @ W_hh^T (K=1024)
// n < 2048 (r,z): gbuf = accA+accB+biases ; n >= 2048: split for r*gh_n
// 192 waves (one per n-tile) -> 24 blocks x 8 waves.
// ---------------------------------------------------------------------------
__global__ void k_gru_gemm(const unsigned* __restrict__ pX,
                           const unsigned* __restrict__ pH,
                           const float* __restrict__ Wih,
                           const float* __restrict__ Whh,
                           const float* __restrict__ bih,
                           const float* __restrict__ bhh,
                           float* __restrict__ gbuf,
                           float* __restrict__ ghn) {
  const int wave = blockIdx.x * 8 + (threadIdx.x >> 5);
  const int lane = threadIdx.x & 31;
  const int n0 = wave * 16;

  v8f accA[4] = {};
  for (int kc = 0; kc < 96; ++kc) {
    v16bf b = load_bt_tile(Wih + (size_t)n0 * H3 + kc * 32, H3, lane);
#pragma unroll
    for (int m = 0; m < 4; ++m) {
      v16bf a = load_packed(pX, m * 96 + kc, lane);
      accA[m] = wmma_bf16(a, b, accA[m]);
    }
  }
  v8f accB[4] = {};
  for (int kc = 0; kc < 32; ++kc) {
    v16bf b = load_bt_tile(Whh + (size_t)n0 * HH + kc * 32, HH, lane);
#pragma unroll
    for (int m = 0; m < 4; ++m) {
      v16bf a = load_packed(pH, m * 32 + kc, lane);
      accB[m] = wmma_bf16(a, b, accB[m]);
    }
  }
  const int row0 = (lane >> 4) * 8;
  const int n = n0 + (lane & 15);
  const float bi = bih[n], bh = bhh[n];
  if (n < 2 * HH) {
#pragma unroll
    for (int m = 0; m < 4; ++m)
#pragma unroll
      for (int i = 0; i < 8; ++i)
        gbuf[(size_t)(m * 16 + row0 + i) * H3 + n] =
            accA[m][i] + accB[m][i] + bi + bh;
  } else {
#pragma unroll
    for (int m = 0; m < 4; ++m)
#pragma unroll
      for (int i = 0; i < 8; ++i) {
        gbuf[(size_t)(m * 16 + row0 + i) * H3 + n] = accA[m][i] + bi;
        ghn[(size_t)(m * 16 + row0 + i) * HH + (n - 2 * HH)] = accB[m][i] + bh;
      }
  }
}

// ---------------------------------------------------------------------------
// Kernel 4b: GRU gates -> h_new (ws) and h_new output slot
// ---------------------------------------------------------------------------
__global__ void k_gru_gate(const float* __restrict__ gbuf,
                           const float* __restrict__ ghn,
                           const float* __restrict__ h0,
                           float* __restrict__ hnew,
                           float* __restrict__ out_h) {
  const int idx = blockIdx.x * blockDim.x + threadIdx.x;
  if (idx >= BB * HH) return;
  const int b = idx >> 10, h = idx & (HH - 1);
  const float* g = gbuf + (size_t)b * H3;
  const float r = 1.0f / (1.0f + __expf(-g[h]));
  const float z = 1.0f / (1.0f + __expf(-g[HH + h]));
  const float ng = tanhf(g[2 * HH + h] + r * ghn[idx]);
  const float hn = (1.0f - z) * ng + z * h0[idx];
  hnew[idx] = hn;
  out_h[idx] = hn;
}

// ---------------------------------------------------------------------------
// Kernel 5: logits = h_new @ W_out^T + b_out  (64 x 50257, K=1024)
// 4 m-tiles x 2 n-tiles per wave -> W_out (206 MB, > L2) streamed exactly
// once from HBM: ~9us floor. 1571 waves over 3142 n-tiles (tail clamped).
// ---------------------------------------------------------------------------
__global__ void k_logits(const unsigned* __restrict__ pA,
                         const float* __restrict__ Wout,
                         const float* __restrict__ bout,
                         float* __restrict__ logits) {
  const int wave = blockIdx.x * 8 + (threadIdx.x >> 5);
  if (wave >= 1571) return;
  const int lane = threadIdx.x & 31;
  const int nt0 = wave * 2;
  v8f acc[8] = {};  // [m*2+t]
  for (int kc = 0; kc < 32; ++kc) {
    v16bf a[4];
#pragma unroll
    for (int m = 0; m < 4; ++m) a[m] = load_packed(pA, m * 32 + kc, lane);
#pragma unroll
    for (int t = 0; t < 2; ++t) {
      v16bf b = load_bt_clamp(Wout, HH, (nt0 + t) * 16, kc * 32, VV, lane);
#pragma unroll
      for (int m = 0; m < 4; ++m)
        acc[m * 2 + t] = wmma_bf16(a[m], b, acc[m * 2 + t]);
    }
  }
  const int row0 = (lane >> 4) * 8;
#pragma unroll
  for (int t = 0; t < 2; ++t) {
    const int n = (nt0 + t) * 16 + (lane & 15);
    if (n < VV) {
      const float bo = bout[n];
#pragma unroll
      for (int m = 0; m < 4; ++m)
#pragma unroll
        for (int i = 0; i < 8; ++i)
          logits[(size_t)(m * 16 + row0 + i) * VV + n] = acc[m * 2 + t][i] + bo;
    }
  }
}

// ---------------------------------------------------------------------------
// Kernel 6: log_softmax over V per row, writes final output
// ---------------------------------------------------------------------------
__global__ void k_logsoftmax(const float* __restrict__ logits,
                             float* __restrict__ out) {
  __shared__ float red[256];
  const int b = blockIdx.x, tid = threadIdx.x;
  const float* row = logits + (size_t)b * VV;
  float lm = -3.0e38f;
  for (int v = tid; v < VV; v += 256) lm = fmaxf(lm, row[v]);
  red[tid] = lm;
  __syncthreads();
  for (int s = 128; s > 0; s >>= 1) {
    if (tid < s) red[tid] = fmaxf(red[tid], red[tid + s]);
    __syncthreads();
  }
  const float mx = red[0];
  __syncthreads();
  float ls = 0.0f;
  for (int v = tid; v < VV; v += 256) ls += __expf(row[v] - mx);
  red[tid] = ls;
  __syncthreads();
  for (int s = 128; s > 0; s >>= 1) {
    if (tid < s) red[tid] += red[tid + s];
    __syncthreads();
  }
  const float lse = mx + __logf(red[0]);
  float* orow = out + (size_t)b * VV;
  for (int v = tid; v < VV; v += 256) orow[v] = row[v] - lse;
}

// ---------------------------------------------------------------------------
// Launch
// ---------------------------------------------------------------------------
extern "C" void kernel_launch(void* const* d_in, const int* in_sizes, int n_in,
                              void* d_out, int out_size, void* d_ws,
                              size_t ws_size, hipStream_t stream) {
  (void)in_sizes; (void)n_in; (void)out_size; (void)ws_size;
  const float* input  = (const float*)d_in[0];   // (B,1,E)
  const float* h0     = (const float*)d_in[1];   // (1,B,H) -> (B,H)
  const float* lch    = (const float*)d_in[2];   // (C,B,H)
  const float* ch     = (const float*)d_in[3];   // (B,C,H)
  const float* topic  = (const float*)d_in[4];   // (B,1,H)
  const unsigned char* pad = (const unsigned char*)d_in[5];  // (B,C) bool
  const float* dynaW  = (const float*)d_in[6];   // (H,H)
  const float* dynaU  = (const float*)d_in[7];   // (H,H)
  const float* dynaV  = (const float*)d_in[8];   // (H,1)
  const float* Wih    = (const float*)d_in[9];   // (3H, 3072)
  const float* Whh    = (const float*)d_in[10];  // (3H, H)
  const float* bih    = (const float*)d_in[11];  // (3H)
  const float* bhh    = (const float*)d_in[12];  // (3H)
  const float* Wout   = (const float*)d_in[13];  // (V, H)
  const float* bout   = (const float*)d_in[14];  // (V)

  // ---- Workspace layout (4-byte words), ~45.8 MB total -------------------
  float* ws      = (float*)d_ws;
  float* q       = ws;                     // B*H   = 65536
  float* scores  = q       + BB * HH;      // B*C   = 12800
  float* applied = scores  + BB * CC;      // B*H   = 65536
  float* gbuf    = applied + BB * HH;      // B*3H  = 196608
  float* ghn     = gbuf    + BB * H3;      // B*H   = 65536
  float* hnew    = ghn     + BB * HH;      // B*H   = 65536
  float* logits  = hnew    + BB * HH;      // B*V   = 3216448
  unsigned* pA_h0   = (unsigned*)(logits + (size_t)BB * VV);
  unsigned* pA_lch  = pA_h0   + (size_t)4   * 32 * 256;  // 32768
  unsigned* pA_x    = pA_lch  + (size_t)800 * 32 * 256;  // 6553600
  unsigned* pA_hnew = pA_x    + (size_t)4   * 96 * 256;  // 98304
  unsigned* pB_W    = pA_hnew + (size_t)4   * 32 * 256;  // 32768
  unsigned* pB_U    = pB_W    + (size_t)64  * 32 * 256;  // 524288
  //           end = pB_U    + 524288

  float* out_logp = (float*)d_out;                 // (B, V)
  float* out_h    = out_logp + (size_t)BB * VV;    // (1, B, H)

  // ---- Pack passes (one-shot, tiny vs. 370 MB of GEMM traffic) ----------
  k_pack_a<<<16, 256, 0, stream>>>(h0, 32, pA_h0);          // 128 tiles
  k_pack_a<<<3200, 256, 0, stream>>>(lch, 32, pA_lch);      // 25600 tiles
  k_pack_bkn<<<256, 256, 0, stream>>>(dynaW, 32, HH, pB_W); // 2048 tiles
  k_pack_bkn<<<256, 256, 0, stream>>>(dynaU, 32, HH, pB_U); // 2048 tiles

  // ---- Attention ---------------------------------------------------------
  k_gemm_q<<<8, 256, 0, stream>>>(pA_h0, pB_W, q);
  k_scores<<<CC * 4, 256, 0, stream>>>(pA_lch, pB_U, q, dynaV, pad, scores);
  k_attn_apply<<<BB, 256, 0, stream>>>(scores, ch, applied);

  // ---- GRU ---------------------------------------------------------------
  k_pack_x<<<48, 256, 0, stream>>>(input, topic, applied, pA_x);  // 384 tiles
  k_gru_gemm<<<24, 256, 0, stream>>>(pA_x, pA_h0, Wih, Whh, bih, bhh, gbuf,
                                     ghn);
  k_gru_gate<<<(BB * HH + 255) / 256, 256, 0, stream>>>(gbuf, ghn, h0, hnew,
                                                        out_h);

  // ---- Output projection + log_softmax -----------------------------------
  k_pack_a<<<16, 256, 0, stream>>>(hnew, 32, pA_hnew);      // 128 tiles
  k_logits<<<197, 256, 0, stream>>>(pA_hnew, Wout, bout, logits);
  k_logsoftmax<<<BB, 256, 0, stream>>>(logits, out_logp);
}
